// Ex4bitLinear_48249662603457
// MI455X (gfx1250) — compile-verified, matrix operand order checked by reading, not run on hardware
//
#include <hip/hip_runtime.h>

// ---------------------------------------------------------------------------
// GPTQ 4-bit dequant + GEMM for MI455X (gfx1250, wave32, WMMA).
//   out[M,N] = x[M,K] * W[K,N],  W = scales[g,n]*(q[k,n] - z[g,n] - 1)
//   M = 8192, K = 4096, N = 11008, group = 128 (g_idx == k/128).
// Compute-bound (739 GFLOP vs ~22us HBM) -> all MACs on v_wmma_f32_16x16x32_f16.
// Software pipeline: global->regs for stage t+1 issued while stage t computes;
// dequant+ds_store at top of next iteration; double-buffered LDS, 1 barrier/stage.
// ---------------------------------------------------------------------------

#define IN_F      4096
#define OUT_F     11008
#define GROUPSIZE 128

#define BM 128
#define BN 128
#define BK 64          // two wmma k-steps (k=32) per LDS stage; 64 stages

typedef __attribute__((ext_vector_type(16))) _Float16 v16h;
typedef __attribute__((ext_vector_type(8)))  _Float16 v8h;
typedef __attribute__((ext_vector_type(4)))  _Float16 v4h;
typedef __attribute__((ext_vector_type(8)))  float    v8f;

__global__ __launch_bounds__(256)
void gptq_gemm_wmma_f16(const float* __restrict__ x,
                        const int*   __restrict__ qweight,   // [K/8, N] nibbles along K
                        const float* __restrict__ scales,    // [K/128, N]
                        const int*   __restrict__ qzeros,    // [K/128, N/8] nibbles along N
                        float*       __restrict__ out)
{
    // A: row-major [m][k] f16; B: column-major [n][k] f16 so one int32's
    // 8 k-adjacent nibbles store as a single ds_store_b128. Double buffered.
    __shared__ _Float16 As[2][BM][BK];   // 2 x 16 KB
    __shared__ _Float16 Bs[2][BN][BK];   // 2 x 16 KB

    const int tid   = threadIdx.x;
    const int lane  = tid & 31;
    const int wave  = tid >> 5;       // 0..7
    const int waveM = wave >> 2;      // 0..1 -> 64-row strip
    const int waveN = wave & 3;       // 0..3 -> 32-col strip
    const int l16   = lane & 15;
    const int hi    = lane >> 4;      // ISA 16-bit frag layout: K-half select

    const int m0 = blockIdx.y * BM;
    const int n0 = blockIdx.x * BN;

    const int c4 = tid & 15;          // float4 column within k-tile (A staging)
    const int r0 = tid >> 4;          // base row (A staging)

    v8f acc[4][2];
#pragma unroll
    for (int mi = 0; mi < 4; ++mi)
#pragma unroll
        for (int ni = 0; ni < 2; ++ni)
            acc[mi][ni] = (v8f){0.f,0.f,0.f,0.f,0.f,0.f,0.f,0.f};

    // ---- register staging for the global->LDS pipeline -----------------
    float4 ar[8];      // A: 128x64 f32 tile, 8 float4 per thread
    int    qw[4];      // B: 8x128 packed int32 tile, 4 per thread
    float  sc[4];
    int    zw[4];

    auto load_stage = [&](int kt) {
#pragma unroll
        for (int p = 0; p < 8; ++p)
            ar[p] = *(const float4*)(x + (size_t)(m0 + r0 + p * 16) * IN_F
                                       + kt + c4 * 4);
        const int g   = kt / GROUPSIZE;   // BK=64 never crosses a group
        const int qk0 = kt >> 3;
#pragma unroll
        for (int p = 0; p < 4; ++p) {
            const int idx = tid + p * 256;
            const int r8  = idx >> 7;     // 0..7
            const int c   = idx & 127;
            const int n   = n0 + c;
            qw[p] = qweight[(size_t)(qk0 + r8) * OUT_F + n];
            sc[p] = scales[(size_t)g * OUT_F + n];
            zw[p] = qzeros[(size_t)g * (OUT_F / 8) + (n >> 3)];
        }
    };

    auto store_stage = [&](int buf) {
#pragma unroll
        for (int p = 0; p < 8; ++p) {
            v4h h;
            h[0] = (_Float16)ar[p].x; h[1] = (_Float16)ar[p].y;
            h[2] = (_Float16)ar[p].z; h[3] = (_Float16)ar[p].w;
            *(v4h*)&As[buf][r0 + p * 16][c4 * 4] = h;        // ds_store_b64
        }
#pragma unroll
        for (int p = 0; p < 4; ++p) {
            const int idx = tid + p * 256;
            const int r8  = idx >> 7;
            const int c   = idx & 127;
            const int n   = n0 + c;
            const float s = sc[p];
            const float z = (float)(((zw[p] >> ((n & 7) * 4)) & 0xF) + 1);
            const float bias = -s * z;                       // w = s*q + bias
            v8h w;
#pragma unroll
            for (int j = 0; j < 8; ++j) {
                const float q = (float)((qw[p] >> (j * 4)) & 0xF);
                w[j] = (_Float16)(s * q + bias);
            }
            *(v8h*)&Bs[buf][c][r8 * 8] = w;                  // ds_store_b128
        }
    };

    auto compute = [&](int buf) {
#pragma unroll
        for (int ks = 0; ks < BK; ks += 32) {
            // lanes 0-15 hold K {0..7,16..23}; lanes 16-31 K {8..15,24..31}
            const int ka = ks + hi * 8;
            v16h a[4], b[2];
#pragma unroll
            for (int mi = 0; mi < 4; ++mi) {
                const int row = waveM * 64 + mi * 16 + l16;
                const v8h lo = *(const v8h*)&As[buf][row][ka];
                const v8h hh = *(const v8h*)&As[buf][row][ka + 16];
                a[mi] = __builtin_shufflevector(lo, hh,
                        0,1,2,3,4,5,6,7,8,9,10,11,12,13,14,15);
            }
#pragma unroll
            for (int ni = 0; ni < 2; ++ni) {
                const int col = waveN * 32 + ni * 16 + l16;
                const v8h lo = *(const v8h*)&Bs[buf][col][ka];
                const v8h hh = *(const v8h*)&Bs[buf][col][ka + 16];
                b[ni] = __builtin_shufflevector(lo, hh,
                        0,1,2,3,4,5,6,7,8,9,10,11,12,13,14,15);
            }
#pragma unroll
            for (int mi = 0; mi < 4; ++mi)
#pragma unroll
                for (int ni = 0; ni < 2; ++ni)
                    acc[mi][ni] = __builtin_amdgcn_wmma_f32_16x16x32_f16(
                        false, a[mi], false, b[ni],
                        (short)0, acc[mi][ni], false, false);
        }
    };

    // ---- pipelined main loop: 64 stages, unrolled x2 for static buf idx ----
    load_stage(0);
    for (int kt = 0; kt < IN_F; kt += 2 * BK) {
        store_stage(0);
        __syncthreads();                 // waits dscnt first -> WAR-safe
        if (kt + BK < IN_F) load_stage(kt + BK);       // overlap with compute
        compute(0);

        store_stage(1);
        __syncthreads();
        if (kt + 2 * BK < IN_F) load_stage(kt + 2 * BK);
        compute(1);
    }

    // ---- epilogue: f32 16x16 C layout (VGPR i -> M=i / i+8; lane -> N) ----
#pragma unroll
    for (int mi = 0; mi < 4; ++mi) {
#pragma unroll
        for (int ni = 0; ni < 2; ++ni) {
            const int col  = n0 + waveN * 32 + ni * 16 + l16;
            const int rowb = m0 + waveM * 64 + mi * 16 + hi * 8;
#pragma unroll
            for (int i = 0; i < 8; ++i)
                out[(size_t)(rowb + i) * OUT_F + col] = acc[mi][ni][i];
        }
    }
}

extern "C" void kernel_launch(void* const* d_in, const int* in_sizes, int n_in,
                              void* d_out, int out_size, void* d_ws, size_t ws_size,
                              hipStream_t stream) {
    const float* x       = (const float*)d_in[0];
    const int*   qweight = (const int*)  d_in[1];
    const float* scales  = (const float*)d_in[2];
    const int*   qzeros  = (const int*)  d_in[3];
    // d_in[4] (g_idx) unused: groups are contiguous (g = k/128) by construction.
    float* out = (float*)d_out;

    const int M = in_sizes[0] / IN_F;            // 8192
    dim3 grid(OUT_F / BN, M / BM);               // (86, 64), exact tiling
    dim3 block(256);                             // 8 wave32s
    hipLaunchKernelGGL(gptq_gemm_wmma_f16, grid, block, 0, stream,
                       x, qweight, scales, qzeros, out);
}
